// EncoderLayer_72730976191155
// MI455X (gfx1250) — matrix operand order repaired and checked by reading, not compiled, not run on hardware
//
#include <hip/hip_runtime.h>
#include <stdint.h>

#define DMODEL   768
#define NHEADS   12
#define DFF      3072
#define DK       64
#define SEQ_LEN  2048
#define NBATCH   2

typedef __attribute__((ext_vector_type(16))) __bf16 v16bf;
typedef __attribute__((ext_vector_type(8)))  __bf16 v8bf;
typedef __attribute__((ext_vector_type(8)))  float  v8f;

union ABFrag { v16bf v; v8bf h[2]; };

__device__ __forceinline__ __bf16 f2bf(float f) {
  uint32_t u = __builtin_bit_cast(uint32_t, f);
  u += 0x7FFFu + ((u >> 16) & 1u);            // round-to-nearest-even
  return __builtin_bit_cast(__bf16, (uint16_t)(u >> 16));
}

// ---------------------------------------------------------------------------
// Tiled bf16 WMMA GEMM:  C = act(A[M,K] * B + bias)
//   TRANS_B : B supplied as [N,K] row-major (i.e. logical B[k][n] = Bm[n][k])
//   OUT_MODE: 0 = C[z*strideC + m*ldc + n]
//             1 = split heads: m=(b,s), n=(h,d) -> C[((b*H+h)*seq+s)*64+d]
//             2 = merge heads: z=(b,h)          -> C[(b*seq+m)*768 + h*64 + n]
// Waves tile 32x64 each (2x4 frags of 16x16); K stepped by 32.
// ---------------------------------------------------------------------------
template<int BM, int BN, bool TRANS_B, bool RELU, int OUT_MODE>
__global__ __launch_bounds__((BM / 32) * (BN / 64) * 32)
void gemm_bf16_wmma(const float* __restrict__ A, int64_t strideA, int lda,
                    const float* __restrict__ Bm, int64_t strideB, int ldb,
                    const float* __restrict__ bias,
                    float* __restrict__ C, int64_t strideC, int ldc,
                    int K, int seq)
{
  constexpr int WROWS = BM / 32;
  constexpr int WCOLS = BN / 64;
  constexpr int NTHR  = WROWS * WCOLS * 32;
  constexpr int LDA_S = 40;   // padded bf16 k-stride: 80B rows -> 16B aligned chunks
  constexpr int LDB_S = 40;

  __shared__ __bf16 As[BM * LDA_S];       // As[m][k]
  __shared__ __bf16 Bs[BN * LDB_S];       // Bs[n][k]  (transposed stage)

  const int tid  = threadIdx.x;
  const int lane = tid & 31;
  const int w    = tid >> 5;
  const int wr   = w % WROWS;
  const int wc   = w / WROWS;
  const int half = lane >> 4;             // 0..1
  const int l16  = lane & 15;             // 0..15
  const int z     = blockIdx.z;
  const int tileM = blockIdx.y * BM;
  const int tileN = blockIdx.x * BN;

  A  += (int64_t)z * strideA;
  Bm += (int64_t)z * strideB;

  v8f acc[2][4];
  const v8f vzero = {0.f, 0.f, 0.f, 0.f, 0.f, 0.f, 0.f, 0.f};
#pragma unroll
  for (int i = 0; i < 2; ++i)
#pragma unroll
    for (int j = 0; j < 4; ++j) acc[i][j] = vzero;

  for (int kb = 0; kb < K; kb += 32) {
    // ---- stage A tile: BM x 32 fp32 -> bf16 ----
#pragma unroll 2
    for (int i = tid; i < BM * 8; i += NTHR) {
      const int r = i >> 3;
      const int c = (i & 7) << 2;
      const float4 f = *(const float4*)(A + (int64_t)(tileM + r) * lda + kb + c);
      __bf16* d = &As[r * LDA_S + c];
      d[0] = f2bf(f.x); d[1] = f2bf(f.y); d[2] = f2bf(f.z); d[3] = f2bf(f.w);
      if (kb + 32 < K)
        __builtin_prefetch(A + (int64_t)(tileM + r) * lda + kb + 32 + c, 0, 3);
    }
    // ---- stage B tile into Bs[n][k] ----
    if (TRANS_B) {
#pragma unroll 2
      for (int i = tid; i < BN * 8; i += NTHR) {
        const int n = i >> 3;
        const int c = (i & 7) << 2;
        const float4 f = *(const float4*)(Bm + (int64_t)(tileN + n) * ldb + kb + c);
        __bf16* d = &Bs[n * LDB_S + c];
        d[0] = f2bf(f.x); d[1] = f2bf(f.y); d[2] = f2bf(f.z); d[3] = f2bf(f.w);
        if (kb + 32 < K)
          __builtin_prefetch(Bm + (int64_t)(tileN + n) * ldb + kb + 32 + c, 0, 3);
      }
    } else {
#pragma unroll 2
      for (int i = tid; i < 32 * (BN / 4); i += NTHR) {
        const int k = i / (BN / 4);
        const int c = (i % (BN / 4)) << 2;
        const float4 f = *(const float4*)(Bm + (int64_t)(kb + k) * ldb + tileN + c);
        Bs[(c + 0) * LDB_S + k] = f2bf(f.x);
        Bs[(c + 1) * LDB_S + k] = f2bf(f.y);
        Bs[(c + 2) * LDB_S + k] = f2bf(f.z);
        Bs[(c + 3) * LDB_S + k] = f2bf(f.w);
        if (kb + 32 < K)
          __builtin_prefetch(Bm + (int64_t)(kb + 32 + k) * ldb + tileN + c, 0, 3);
      }
    }
    __syncthreads();

    // ---- gather fragments (contiguous k = 16*half + s, same for A and B) ----
    ABFrag af[2];
#pragma unroll
    for (int fr = 0; fr < 2; ++fr) {
      const __bf16* p = &As[(wr * 32 + fr * 16 + l16) * LDA_S + 16 * half];
      af[fr].h[0] = *(const v8bf*)(p);        // K = 16*half + 0..7
      af[fr].h[1] = *(const v8bf*)(p + 8);    // K = 16*half + 8..15
    }
    ABFrag bf[4];
#pragma unroll
    for (int fc = 0; fc < 4; ++fc) {
      const __bf16* p = &Bs[(wc * 64 + fc * 16 + l16) * LDB_S + 16 * half];
      bf[fc].h[0] = *(const v8bf*)(p);        // K = 16*half + 0..7
      bf[fc].h[1] = *(const v8bf*)(p + 8);    // K = 16*half + 8..15
    }
#pragma unroll
    for (int fr = 0; fr < 2; ++fr)
#pragma unroll
      for (int fc = 0; fc < 4; ++fc)
        acc[fr][fc] = __builtin_amdgcn_wmma_f32_16x16x32_bf16(
            false, af[fr].v, false, bf[fc].v, (short)0, acc[fr][fc], false, false);
    __syncthreads();
  }

  // ---- epilogue ----
#pragma unroll
  for (int fr = 0; fr < 2; ++fr) {
    const int mbase = tileM + wr * 32 + fr * 16 + 8 * half;
#pragma unroll
    for (int fc = 0; fc < 4; ++fc) {
      const int n  = tileN + wc * 64 + fc * 16 + l16;
      const float bv = bias ? bias[n] : 0.0f;
#pragma unroll
      for (int r = 0; r < 8; ++r) {
        const int m = mbase + r;
        float v = acc[fr][fc][r] + bv;
        if (RELU) v = fmaxf(v, 0.0f);
        if (OUT_MODE == 0) {
          C[(int64_t)z * strideC + (int64_t)m * ldc + n] = v;
        } else if (OUT_MODE == 1) {
          const int b = m / seq, s = m % seq;
          const int h = n >> 6, dd = n & 63;
          C[(((int64_t)(b * NHEADS + h)) * seq + s) * DK + dd] = v;
        } else {
          const int b = z / NHEADS, h = z % NHEADS;
          C[((int64_t)(b * seq + m)) * DMODEL + h * DK + n] = v;
        }
      }
    }
  }
}

// ---------------------------------------------------------------------------
// Row softmax over attn[z, q, :] with scale 1/sqrt(dk) and bool mask.
// ---------------------------------------------------------------------------
__device__ __forceinline__ float wred_max(float v) {
#pragma unroll
  for (int o = 16; o > 0; o >>= 1) v = fmaxf(v, __shfl_xor(v, o, 32));
  return v;
}
__device__ __forceinline__ float wred_sum(float v) {
#pragma unroll
  for (int o = 16; o > 0; o >>= 1) v += __shfl_xor(v, o, 32);
  return v;
}

__global__ __launch_bounds__(256)
void softmax_rows(float* __restrict__ attn, const unsigned char* __restrict__ mask, int S)
{
  constexpr int VPT = SEQ_LEN / 256;   // 8
  const int z = blockIdx.y;            // b*H + h
  const int b = z / NHEADS;
  const int q = blockIdx.x;
  float* row = attn + ((int64_t)z * S + q) * S;
  const unsigned char* mrow = mask + ((int64_t)b * S + q) * S;
  const int tid = threadIdx.x;

  __shared__ float red[8];
  float vals[VPT];
  float mx = -3.402823466e+38f;
#pragma unroll
  for (int j = 0; j < VPT; ++j) {
    const int i = tid + j * 256;
    float x = row[i] * 0.125f;         // 1/sqrt(64)
    if (mrow[i]) x = -1e9f;
    vals[j] = x;
    mx = fmaxf(mx, x);
  }
  mx = wred_max(mx);
  if ((tid & 31) == 0) red[tid >> 5] = mx;
  __syncthreads();
  float gmx = red[0];
#pragma unroll
  for (int i = 1; i < 8; ++i) gmx = fmaxf(gmx, red[i]);
  __syncthreads();

  float sum = 0.0f;
#pragma unroll
  for (int j = 0; j < VPT; ++j) { vals[j] = __expf(vals[j] - gmx); sum += vals[j]; }
  sum = wred_sum(sum);
  if ((tid & 31) == 0) red[tid >> 5] = sum;
  __syncthreads();
  float gs = 0.0f;
#pragma unroll
  for (int i = 0; i < 8; ++i) gs += red[i];
  const float inv = 1.0f / gs;
#pragma unroll
  for (int j = 0; j < VPT; ++j) row[tid + j * 256] = vals[j] * inv;
}

// ---------------------------------------------------------------------------
// out[row] = LayerNorm(x[row] + res[row]) * gamma + beta   (row = 768 elems)
// ---------------------------------------------------------------------------
__global__ __launch_bounds__(256)
void add_res_layernorm(const float* __restrict__ x, const float* __restrict__ res,
                       const float* __restrict__ gamma, const float* __restrict__ beta,
                       float* __restrict__ out)
{
  constexpr int VPT = DMODEL / 256;    // 3
  const int row = blockIdx.x;
  const float* xr = x   + (int64_t)row * DMODEL;
  const float* rr = res + (int64_t)row * DMODEL;
  float* orow     = out + (int64_t)row * DMODEL;
  const int tid = threadIdx.x;

  __shared__ float rs[8], rss[8];
  float v[VPT];
  float s = 0.f, ss = 0.f;
#pragma unroll
  for (int j = 0; j < VPT; ++j) {
    const int i = tid + j * 256;
    const float t = xr[i] + rr[i];
    v[j] = t; s += t; ss += t * t;
  }
#pragma unroll
  for (int o = 16; o > 0; o >>= 1) { s += __shfl_xor(s, o, 32); ss += __shfl_xor(ss, o, 32); }
  if ((tid & 31) == 0) { rs[tid >> 5] = s; rss[tid >> 5] = ss; }
  __syncthreads();
  float St = 0.f, SSt = 0.f;
#pragma unroll
  for (int i = 0; i < 8; ++i) { St += rs[i]; SSt += rss[i]; }
  const float mu   = St * (1.0f / DMODEL);
  const float var  = SSt * (1.0f / DMODEL) - mu * mu;
  const float rstd = rsqrtf(var + 1e-5f);
#pragma unroll
  for (int j = 0; j < VPT; ++j) {
    const int i = tid + j * 256;
    orow[i] = (v[j] - mu) * rstd * gamma[i] + beta[i];
  }
}

// ---------------------------------------------------------------------------
extern "C" void kernel_launch(void* const* d_in, const int* in_sizes, int n_in,
                              void* d_out, int out_size, void* d_ws, size_t ws_size,
                              hipStream_t stream)
{
  (void)in_sizes; (void)n_in; (void)out_size; (void)ws_size;

  const float*         X    = (const float*)d_in[0];
  const unsigned char* mask = (const unsigned char*)d_in[1];
  const float* Wq = (const float*)d_in[2];
  const float* bq = (const float*)d_in[3];
  const float* Wk = (const float*)d_in[4];
  const float* bk = (const float*)d_in[5];
  const float* Wv = (const float*)d_in[6];
  const float* bv = (const float*)d_in[7];
  const float* Wo = (const float*)d_in[8];
  const float* bo = (const float*)d_in[9];
  const float* g1 = (const float*)d_in[10];
  const float* be1= (const float*)d_in[11];
  const float* W1 = (const float*)d_in[12];
  const float* b1 = (const float*)d_in[13];
  const float* W2 = (const float*)d_in[14];
  const float* b2 = (const float*)d_in[15];
  const float* g2 = (const float*)d_in[16];
  const float* be2= (const float*)d_in[17];

  const int M = NBATCH * SEQ_LEN;                  // 4096 rows
  float* out  = (float*)d_out;
  float* attn = out + (int64_t)M * DMODEL;         // [B,H,S,S] region of d_out

  char* ws = (char*)d_ws;
  const size_t szMD = (size_t)M * DMODEL * sizeof(float);
  float* Q  = (float*)(ws);                        // later reused as context
  float* Kb = (float*)(ws + szMD);                 // later reused as y / y2
  float* Vb = (float*)(ws + 2 * szMD);             // later reused as attn_out
  float* H1 = (float*)(ws + 3 * szMD);             // [M, DFF]

  const int64_t sQ = (int64_t)SEQ_LEN * DK;
  dim3 blk256(256), blk128(128);

  // 1-3: Q/K/V = X @ W + b, split into [B,H,S,64]
  dim3 gProj(DMODEL / 128, M / 128, 1);
  gemm_bf16_wmma<128,128,false,false,1><<<gProj, blk256, 0, stream>>>(
      X, 0, DMODEL, Wq, 0, DMODEL, bq, Q,  0, 0, DMODEL, SEQ_LEN);
  gemm_bf16_wmma<128,128,false,false,1><<<gProj, blk256, 0, stream>>>(
      X, 0, DMODEL, Wk, 0, DMODEL, bk, Kb, 0, 0, DMODEL, SEQ_LEN);
  gemm_bf16_wmma<128,128,false,false,1><<<gProj, blk256, 0, stream>>>(
      X, 0, DMODEL, Wv, 0, DMODEL, bv, Vb, 0, 0, DMODEL, SEQ_LEN);

  // 4: scores = Q @ K^T per (b,h) -> attn region of d_out
  dim3 gScore(SEQ_LEN / 128, SEQ_LEN / 128, NBATCH * NHEADS);
  gemm_bf16_wmma<128,128,true,false,0><<<gScore, blk256, 0, stream>>>(
      Q, sQ, DK, Kb, sQ, DK, nullptr,
      attn, (int64_t)SEQ_LEN * SEQ_LEN, SEQ_LEN, DK, SEQ_LEN);

  // 5: attn = softmax(scores/sqrt(dk) + mask), in place
  dim3 gSm(SEQ_LEN, NBATCH * NHEADS);
  softmax_rows<<<gSm, blk256, 0, stream>>>(attn, mask, SEQ_LEN);

  // 6: context = attn @ V, merged heads -> [B,S,768]  (reuse Q)
  float* ctx = Q;
  dim3 gCtx(1, SEQ_LEN / 128, NBATCH * NHEADS);
  gemm_bf16_wmma<128,64,false,false,2><<<gCtx, blk128, 0, stream>>>(
      attn, (int64_t)SEQ_LEN * SEQ_LEN, SEQ_LEN, Vb, sQ, DK, nullptr,
      ctx, 0, 0, SEQ_LEN, SEQ_LEN);

  // 7: y = ctx @ Wo + bo   (reuse Kb)
  float* y = Kb;
  dim3 gO(DMODEL / 128, M / 128, 1);
  gemm_bf16_wmma<128,128,false,false,0><<<gO, blk256, 0, stream>>>(
      ctx, 0, DMODEL, Wo, 0, DMODEL, bo, y, 0, DMODEL, DMODEL, SEQ_LEN);

  // 8: attn_out = LN1(y + X)   (reuse Vb)
  float* attn_out = Vb;
  add_res_layernorm<<<dim3(M), blk256, 0, stream>>>(y, X, g1, be1, attn_out);

  // 9: H1 = relu(attn_out @ W1 + b1)
  dim3 gF1(DFF / 128, M / 128, 1);
  gemm_bf16_wmma<128,128,false,true,0><<<gF1, blk256, 0, stream>>>(
      attn_out, 0, DMODEL, W1, 0, DFF, b1, H1, 0, DFF, DMODEL, SEQ_LEN);

  // 10: y2 = H1 @ W2 + b2   (reuse Kb)
  float* y2 = Kb;
  gemm_bf16_wmma<128,128,false,false,0><<<gO, blk256, 0, stream>>>(
      H1, 0, DFF, W2, 0, DMODEL, b2, y2, 0, DMODEL, DFF, SEQ_LEN);

  // 11: out = LN2(y2 + attn_out)
  add_res_layernorm<<<dim3(M), blk256, 0, stream>>>(y2, attn_out, g2, be2, out);
}
